// AttentionLayer_37615323578943
// MI455X (gfx1250) — matrix operand order repaired and checked by reading, not compiled, not run on hardware
//
#include <hip/hip_runtime.h>

typedef __attribute__((ext_vector_type(16))) _Float16 v16h;
typedef __attribute__((ext_vector_type(8)))  _Float16 v8h;
typedef __attribute__((ext_vector_type(4)))  _Float16 v4h;
typedef __attribute__((ext_vector_type(8)))  float    v8f;
typedef __attribute__((ext_vector_type(4)))  float    v4f;
typedef __attribute__((ext_vector_type(4)))  unsigned int u32x4;
typedef __attribute__((ext_vector_type(4)))  int      i32x4;
typedef __attribute__((ext_vector_type(8)))  int      i32x8;

#define GAMMA 1.0f

constexpr int B_ = 4;
constexpr int N_ = 2048;
constexpr int M_ = 2048;
constexpr int D_ = 128;
constexpr int QB = 128;           // Q rows per workgroup (16 per wave, 8 waves)
constexpr int JT = 64;            // K/V rows per tile iteration
constexpr int SPLITS = 4;         // split-K over the key dimension
constexpr int MS = M_ / SPLITS;   // keys per split (512)
constexpr float LOG2E = 1.4426950408889634f;

// Workspace layout (bytes)
constexpr size_t KH_OFF   = 0;                                    // f16 K  [b][j][d]
constexpr size_t VT_OFF   = (size_t)B_ * M_ * D_ * 2;             // f16 Vt [b][d][j]
constexpr size_t BIAS_OFF = VT_OFF + (size_t)B_ * M_ * D_ * 2;    // f32 -g*log2e*||k||^2 [b][j]
constexpr size_t OP_OFF   = BIAS_OFF + (size_t)B_ * M_ * 4;       // f32 partial O [s][b][n][d]
constexpr size_t ML_OFF   = OP_OFF + (size_t)SPLITS * B_ * N_ * D_ * 4; // f32 (m,l) [s][b][n][2]
constexpr size_t WS_NEED  = ML_OFF + (size_t)SPLITS * B_ * N_ * 2 * 4;

static __device__ inline unsigned lds_byte_addr(const void* p) {
    return (unsigned)(uintptr_t)p;   // low 32 bits of generic LDS pointer = LDS byte address
}

// Issue one 2D TDM tile load (f16 elements): tile_d0 x tile_d1, row stride stride0 (elements).
static __device__ inline void tdm_load_2d_f16(unsigned lds_addr, const void* gptr,
                                              unsigned tensor_d0, unsigned tensor_d1,
                                              unsigned long long stride0,
                                              unsigned tile_d0, unsigned tile_d1)
{
    unsigned long long ga = (unsigned long long)(uintptr_t)gptr;
    u32x4 g0;
    g0[0] = 1u;                                               // count=1, user descriptor
    g0[1] = lds_addr;                                         // LDS dest (bytes)
    g0[2] = (unsigned)(ga & 0xffffffffu);                     // global_addr[31:0]
    g0[3] = (unsigned)((ga >> 32) & 0x1ffffffu) | (2u << 30); // global_addr[56:32] | type=2
    i32x8 g1;
    g1[0] = (int)(1u << 16);                                  // data_size=1 (2 bytes)
    g1[1] = (int)((tensor_d0 & 0xffffu) << 16);
    g1[2] = (int)((tensor_d0 >> 16) | ((tensor_d1 & 0xffffu) << 16));
    g1[3] = (int)((tensor_d1 >> 16) | ((tile_d0 & 0xffffu) << 16));
    g1[4] = (int)(tile_d1 & 0xffffu);                         // tile_dim1 (tile_dim2=0)
    g1[5] = (int)(unsigned)(stride0 & 0xffffffffu);
    g1[6] = (int)(unsigned)((stride0 >> 32) & 0xffffu);
    g1[7] = 0;
    i32x4 z4 = (i32x4)0;
#if __clang_major__ >= 23
    i32x8 z8 = (i32x8)0;
    __builtin_amdgcn_tensor_load_to_lds(g0, g1, z4, z4, z8, 0);
#else
    __builtin_amdgcn_tensor_load_to_lds(g0, g1, z4, z4, 0);
#endif
}

// ---------------------------------------------------------------------------
// Preprocess: K -> f16 [b][j][d], V -> f16 transposed [b][d][j],
//             bias = -gamma*log2e*||k_j||^2
// ---------------------------------------------------------------------------
__global__ __launch_bounds__(256)
void rbf_prep_kernel(const float* __restrict__ k, const float* __restrict__ v,
                     char* __restrict__ ws)
{
    __shared__ _Float16 sv[JT * D_];   // staged V tile, row-major [j][d]

    _Float16* kh  = (_Float16*)(ws + KH_OFF);
    _Float16* vtw = (_Float16*)(ws + VT_OFF);
    float*    bsg = (float*)(ws + BIAS_OFF);

    const int tid = threadIdx.x;
    const int blk = blockIdx.x;
    const int b   = blk / (M_ / JT);
    const int j0  = (blk % (M_ / JT)) * JT;

    for (int it = 0; it < 8; ++it) {
        int lin = it * 256 + tid;          // 0..2047 float4 slots
        int j   = lin >> 5;
        int d   = (lin & 31) * 4;
        v4f kv = *(const v4f*)(k + ((size_t)b * M_ + j0 + j) * (size_t)D_ + d);
        v4f vv = *(const v4f*)(v + ((size_t)b * M_ + j0 + j) * (size_t)D_ + d);
        v4h khh, vhh;
        for (int e = 0; e < 4; ++e) { khh[e] = (_Float16)kv[e]; vhh[e] = (_Float16)vv[e]; }
        *(v4h*)&kh[((size_t)b * M_ + j0 + j) * (size_t)D_ + d] = khh;
        *(v4h*)&sv[j * D_ + d] = vhh;
    }
    if (tid < JT) {
        const float* kr = k + ((size_t)b * M_ + j0 + tid) * (size_t)D_;
        float s2 = 0.0f;
        for (int d = 0; d < D_; d += 4) {
            v4f x = *(const v4f*)(kr + d);
            s2 += x[0]*x[0] + x[1]*x[1] + x[2]*x[2] + x[3]*x[3];
        }
        bsg[(size_t)b * M_ + j0 + tid] = -GAMMA * LOG2E * s2;
    }
    __syncthreads();

    // Write transposed V: 512 chunks of 16 contiguous j's
    for (int it = 0; it < 2; ++it) {
        int idx  = it * 256 + tid;         // 0..511
        int d    = idx >> 2;
        int part = idx & 3;
        _Float16 tmp[16] __attribute__((aligned(16)));
        for (int jj = 0; jj < 16; ++jj) tmp[jj] = sv[(part * 16 + jj) * D_ + d];
        _Float16* dst = vtw + (size_t)b * D_ * M_ + (size_t)d * M_ + j0 + part * 16;
        *(v8h*)dst       = *(const v8h*)tmp;
        *(v8h*)(dst + 8) = *(const v8h*)(tmp + 8);
    }
}

// ---------------------------------------------------------------------------
// Main kernel: TDM double-buffered split-K flash attention (exp2 domain)
// Grid: SPLITS * B * (N/QB) workgroups of 256 threads.
// ---------------------------------------------------------------------------
__global__ __launch_bounds__(256)
void rbf_attn_tdm_kernel(const float* __restrict__ q, char* __restrict__ ws)
{
    __shared__ _Float16 lds_k[2][JT * D_];   // [j][d] f16
    __shared__ _Float16 lds_vt[2][D_ * JT];  // [d][j] f16
    __shared__ _Float16 lds_p[8][16 * JT];   // per-wave P transpose buffer

    const _Float16* kh  = (const _Float16*)(ws + KH_OFF);
    const _Float16* vtg = (const _Float16*)(ws + VT_OFF);
    const float*    bsg = (const float*)(ws + BIAS_OFF);
    float*          opart = (float*)(ws + OP_OFF);
    float*          mlp   = (float*)(ws + ML_OFF);

    const int tid  = threadIdx.x;
    const int wave = tid >> 5;
    const int lane = tid & 31;
    const int c    = lane & 15;
    const int h16  = (lane >> 4) & 1;
    const int ksa  = h16 * 8;
    const int ksb  = h16 * 8;
    const int mb   = h16 * 8;

    const int nQB = N_ / QB;
    const int blk = blockIdx.x;
    const int s   = blk / (B_ * nQB);           // key-split index
    const int rem = blk % (B_ * nQB);
    const int b   = rem / nQB;
    const int q0  = (rem % nQB) * QB;
    const int qr  = q0 + wave * 16 + c;

    // Q fragments (scaled by 2*gamma*log2e for exp2-domain softmax)
    const float* qrow = q + ((size_t)b * N_ + qr) * (size_t)D_;
    const float qscale = 2.0f * GAMMA * LOG2E;
    v16h qf[4];
    for (int kc = 0; kc < 4; ++kc) {
        const float* p0 = qrow + kc * 32 + ksa;
        const float* p1 = qrow + kc * 32 + 16 + ksa;
        v4f a0 = *(const v4f*)p0, a1 = *(const v4f*)(p0 + 4);
        v4f b0 = *(const v4f*)p1, b1 = *(const v4f*)(p1 + 4);
        for (int e = 0; e < 4; ++e) {
            qf[kc][e]      = (_Float16)(qscale * a0[e]);
            qf[kc][4 + e]  = (_Float16)(qscale * a1[e]);
            qf[kc][8 + e]  = (_Float16)(qscale * b0[e]);
            qf[kc][12 + e] = (_Float16)(qscale * b1[e]);
        }
    }

    v8f o[8];
    for (int i = 0; i < 8; ++i) o[i] = (v8f)0.0f;
    float mrow[8], lrow[8];
    for (int i = 0; i < 8; ++i) { mrow[i] = -3.0e38f; lrow[i] = 0.0f; }

    const int jbase = s * MS;
    const int nt = MS / JT;
    if (wave == 0) {
        tdm_load_2d_f16(lds_byte_addr(&lds_k[0][0]),
                        kh + ((size_t)b * M_ + jbase) * (size_t)D_,
                        D_, JT, D_, D_, JT);
        tdm_load_2d_f16(lds_byte_addr(&lds_vt[0][0]),
                        vtg + (size_t)b * D_ * M_ + jbase,
                        JT, D_, M_, JT, D_);
    }

    for (int t = 0; t < nt; ++t) {
        const int buf = t & 1;
        const int j0  = jbase + t * JT;

        __syncthreads();  // all waves done reading buffer buf^1 from tile t-1
        if (wave == 0) {
            if (t + 1 < nt) {
                const int j1 = jbase + (t + 1) * JT;
                tdm_load_2d_f16(lds_byte_addr(&lds_k[buf ^ 1][0]),
                                kh + ((size_t)b * M_ + j1) * (size_t)D_,
                                D_, JT, D_, D_, JT);
                tdm_load_2d_f16(lds_byte_addr(&lds_vt[buf ^ 1][0]),
                                vtg + (size_t)b * D_ * M_ + j1,
                                JT, D_, M_, JT, D_);
                __builtin_amdgcn_s_wait_tensorcnt(2);   // tile t's two TDM ops done
            } else {
                __builtin_amdgcn_s_wait_tensorcnt(0);
            }
        }
        __syncthreads();  // tile t resident for all waves

        // ---- S = (2g*log2e Q) K^T + bias : 4 N-blocks x 4 K-chunks of WMMA ----
        v8f sc[4];
        for (int n = 0; n < 4; ++n) {
            const int j = n * 16 + c;
            v8f acc = (v8f)0.0f;
            for (int kc = 0; kc < 4; ++kc) {
                const _Float16* bp = &lds_k[buf][j * D_ + kc * 32 + ksb];
                v8h x0 = *(const v8h*)bp;
                v8h x1 = *(const v8h*)(bp + 16);
                v16h bf;
                for (int e = 0; e < 8; ++e) { bf[e] = x0[e]; bf[8 + e] = x1[e]; }
                acc = __builtin_amdgcn_wmma_f32_16x16x32_f16(
                          false, qf[kc], false, bf, (short)0, acc, false, false);
            }
            float bias = bsg[(size_t)b * M_ + j0 + j];
            for (int i = 0; i < 8; ++i) acc[i] += bias;
            sc[n] = acc;
        }

        // ---- Online softmax in exp2 domain ----
        float mnew[8], scale[8];
        for (int i = 0; i < 8; ++i) {
            float tmax = fmaxf(fmaxf(sc[0][i], sc[1][i]), fmaxf(sc[2][i], sc[3][i]));
            for (int off = 1; off < 16; off <<= 1) tmax = fmaxf(tmax, __shfl_xor(tmax, off));
            mnew[i]  = fmaxf(mrow[i], tmax);
            scale[i] = exp2f(mrow[i] - mnew[i]);
            mrow[i]  = mnew[i];
        }
        float rs[8];
        for (int i = 0; i < 8; ++i) rs[i] = 0.0f;
        for (int n = 0; n < 4; ++n)
            for (int i = 0; i < 8; ++i) {
                float p = exp2f(sc[n][i] - mnew[i]);
                sc[n][i] = p;
                rs[i] += p;
            }
        for (int i = 0; i < 8; ++i) {
            float tsum = rs[i];
            for (int off = 1; off < 16; off <<= 1) tsum += __shfl_xor(tsum, off);
            lrow[i] = lrow[i] * scale[i] + tsum;
        }
        for (int db = 0; db < 8; ++db)
            for (int i = 0; i < 8; ++i) o[db][i] *= scale[i];

        // ---- Transpose P (C layout -> A layout) via per-wave LDS ----
        _Float16* pw = lds_p[wave];
        for (int n = 0; n < 4; ++n)
            for (int i = 0; i < 8; ++i)
                pw[(mb + i) * JT + n * 16 + c] = (_Float16)sc[n][i];
        __asm__ volatile("s_wait_dscnt 0" ::: "memory");

        v16h pf[2];
        for (int ac = 0; ac < 2; ++ac) {
            const _Float16* pp = &pw[c * JT + ac * 32 + ksa];
            v8h x0 = *(const v8h*)pp;
            v8h x1 = *(const v8h*)(pp + 16);
            for (int e = 0; e < 8; ++e) { pf[ac][e] = x0[e]; pf[ac][8 + e] = x1[e]; }
        }

        // ---- O += P V ----
        for (int db = 0; db < 8; ++db) {
            v8f acc = o[db];
            for (int ac = 0; ac < 2; ++ac) {
                const _Float16* vp = &lds_vt[buf][(db * 16 + c) * JT + ac * 32 + ksb];
                v8h x0 = *(const v8h*)vp;
                v8h x1 = *(const v8h*)(vp + 16);
                v16h vf;
                for (int e = 0; e < 8; ++e) { vf[e] = x0[e]; vf[8 + e] = x1[e]; }
                acc = __builtin_amdgcn_wmma_f32_16x16x32_f16(
                          false, pf[ac], false, vf, (short)0, acc, false, false);
            }
            o[db] = acc;
        }
    }

    // ---- Epilogue: store unnormalized partial O and (m, l) per row ----
    for (int i = 0; i < 8; ++i) {
        const size_t r  = (size_t)b * N_ + (q0 + wave * 16 + mb + i);
        const size_t rr = (size_t)s * B_ * N_ + r;
        float* orow = opart + rr * (size_t)D_;
        for (int db = 0; db < 8; ++db)
            orow[db * 16 + c] = o[db][i];
        if (c == 0) {
            mlp[rr * 2 + 0] = mrow[i];
            mlp[rr * 2 + 1] = lrow[i];
        }
    }
}

// ---------------------------------------------------------------------------
// Combine: merge SPLITS partials per row (exp2 domain)
// ---------------------------------------------------------------------------
__global__ __launch_bounds__(256)
void rbf_combine_kernel(const char* __restrict__ ws, float* __restrict__ out)
{
    const float* opart = (const float*)(ws + OP_OFF);
    const float* mlp   = (const float*)(ws + ML_OFF);

    const size_t idx = (size_t)blockIdx.x * 256 + threadIdx.x;  // over B*N*D
    const size_t r   = idx / D_;
    const size_t d   = idx % D_;

    float m[SPLITS], l[SPLITS];
    float gmax = -3.0e38f;
    for (int ss = 0; ss < SPLITS; ++ss) {
        const size_t rr = (size_t)ss * B_ * N_ + r;
        m[ss] = mlp[rr * 2 + 0];
        l[ss] = mlp[rr * 2 + 1];
        gmax = fmaxf(gmax, m[ss]);
    }
    float num = 0.0f, den = 0.0f;
    for (int ss = 0; ss < SPLITS; ++ss) {
        const size_t rr = (size_t)ss * B_ * N_ + r;
        float w = exp2f(m[ss] - gmax);
        num += w * opart[rr * (size_t)D_ + d];
        den += w * l[ss];
    }
    out[idx] = num / den;
}

// ---------------------------------------------------------------------------
// Fallback (self-contained, no workspace): fused kernel
// ---------------------------------------------------------------------------
__global__ __launch_bounds__(256)
void rbf_attn_fused_kernel(const float* __restrict__ q,
                           const float* __restrict__ k,
                           const float* __restrict__ v,
                           float* __restrict__ out)
{
    __shared__ _Float16 lds_k[JT * D_];
    __shared__ _Float16 lds_vt[D_ * JT];
    __shared__ float    lds_bias[JT];
    __shared__ _Float16 lds_p[8][16 * JT];

    const int tid  = threadIdx.x;
    const int wave = tid >> 5;
    const int lane = tid & 31;
    const int c    = lane & 15;
    const int h16  = (lane >> 4) & 1;
    const int ksa  = h16 * 8;
    const int ksb  = h16 * 8;
    const int mb   = h16 * 8;

    const int blk = blockIdx.x;
    const int b   = blk / (N_ / QB);
    const int q0  = (blk % (N_ / QB)) * QB;
    const int qr  = q0 + wave * 16 + c;

    const float* qrow = q + ((size_t)b * N_ + qr) * (size_t)D_;
    const float qscale = 2.0f * GAMMA * LOG2E;
    v16h qf[4];
    for (int kc = 0; kc < 4; ++kc) {
        const float* p0 = qrow + kc * 32 + ksa;
        const float* p1 = qrow + kc * 32 + 16 + ksa;
        v4f a0 = *(const v4f*)p0, a1 = *(const v4f*)(p0 + 4);
        v4f b0 = *(const v4f*)p1, b1 = *(const v4f*)(p1 + 4);
        for (int e = 0; e < 4; ++e) {
            qf[kc][e]      = (_Float16)(qscale * a0[e]);
            qf[kc][4 + e]  = (_Float16)(qscale * a1[e]);
            qf[kc][8 + e]  = (_Float16)(qscale * b0[e]);
            qf[kc][12 + e] = (_Float16)(qscale * b1[e]);
        }
    }

    v8f o[8];
    for (int i = 0; i < 8; ++i) o[i] = (v8f)0.0f;
    float mrow[8], lrow[8];
    for (int i = 0; i < 8; ++i) { mrow[i] = -3.0e38f; lrow[i] = 0.0f; }

    for (int j0 = 0; j0 < M_; j0 += JT) {
        __syncthreads();
        for (int it = 0; it < 8; ++it) {
            int lin = it * 256 + tid;
            int j   = lin >> 5;
            int d   = (lin & 31) * 4;
            v4f kv = *(const v4f*)(k + ((size_t)b * M_ + j0 + j) * (size_t)D_ + d);
            v4f vv = *(const v4f*)(v + ((size_t)b * M_ + j0 + j) * (size_t)D_ + d);
            v4h khv;
            for (int e = 0; e < 4; ++e) khv[e] = (_Float16)kv[e];
            *(v4h*)&lds_k[j * D_ + d] = khv;
            for (int e = 0; e < 4; ++e) lds_vt[(d + e) * JT + j] = (_Float16)vv[e];
        }
        if (tid < JT) {
            const float* kr = k + ((size_t)b * M_ + j0 + tid) * (size_t)D_;
            float s2 = 0.0f;
            for (int d = 0; d < D_; d += 4) {
                v4f x = *(const v4f*)(kr + d);
                s2 += x[0]*x[0] + x[1]*x[1] + x[2]*x[2] + x[3]*x[3];
            }
            lds_bias[tid] = -GAMMA * LOG2E * s2;
        }
        __syncthreads();

        v8f sc[4];
        for (int n = 0; n < 4; ++n) {
            const int j = n * 16 + c;
            v8f acc = (v8f)0.0f;
            for (int kc = 0; kc < 4; ++kc) {
                const _Float16* bp = &lds_k[j * D_ + kc * 32 + ksb];
                v8h x0 = *(const v8h*)bp;
                v8h x1 = *(const v8h*)(bp + 16);
                v16h bf;
                for (int e = 0; e < 8; ++e) { bf[e] = x0[e]; bf[8 + e] = x1[e]; }
                acc = __builtin_amdgcn_wmma_f32_16x16x32_f16(
                          false, qf[kc], false, bf, (short)0, acc, false, false);
            }
            float bias = lds_bias[j];
            for (int i = 0; i < 8; ++i) acc[i] += bias;
            sc[n] = acc;
        }

        float mnew[8], scale[8];
        for (int i = 0; i < 8; ++i) {
            float tmax = fmaxf(fmaxf(sc[0][i], sc[1][i]), fmaxf(sc[2][i], sc[3][i]));
            for (int off = 1; off < 16; off <<= 1) tmax = fmaxf(tmax, __shfl_xor(tmax, off));
            mnew[i]  = fmaxf(mrow[i], tmax);
            scale[i] = exp2f(mrow[i] - mnew[i]);
            mrow[i]  = mnew[i];
        }
        float rs[8];
        for (int i = 0; i < 8; ++i) rs[i] = 0.0f;
        for (int n = 0; n < 4; ++n)
            for (int i = 0; i < 8; ++i) {
                float p = exp2f(sc[n][i] - mnew[i]);
                sc[n][i] = p;
                rs[i] += p;
            }
        for (int i = 0; i < 8; ++i) {
            float tsum = rs[i];
            for (int off = 1; off < 16; off <<= 1) tsum += __shfl_xor(tsum, off);
            lrow[i] = lrow[i] * scale[i] + tsum;
        }
        for (int db = 0; db < 8; ++db)
            for (int i = 0; i < 8; ++i) o[db][i] *= scale[i];

        _Float16* pw = lds_p[wave];
        for (int n = 0; n < 4; ++n)
            for (int i = 0; i < 8; ++i)
                pw[(mb + i) * JT + n * 16 + c] = (_Float16)sc[n][i];
        __asm__ volatile("s_wait_dscnt 0" ::: "memory");

        v16h pf[2];
        for (int ac = 0; ac < 2; ++ac) {
            const _Float16* pp = &pw[c * JT + ac * 32 + ksa];
            v8h x0 = *(const v8h*)pp;
            v8h x1 = *(const v8h*)(pp + 16);
            for (int e = 0; e < 8; ++e) { pf[ac][e] = x0[e]; pf[ac][8 + e] = x1[e]; }
        }

        for (int db = 0; db < 8; ++db) {
            v8f acc = o[db];
            for (int ac = 0; ac < 2; ++ac) {
                const _Float16* vp = &lds_vt[(db * 16 + c) * JT + ac * 32 + ksb];
                v8h x0 = *(const v8h*)vp;
                v8h x1 = *(const v8h*)(vp + 16);
                v16h vf;
                for (int e = 0; e < 8; ++e) { vf[e] = x0[e]; vf[8 + e] = x1[e]; }
                acc = __builtin_amdgcn_wmma_f32_16x16x32_f16(
                          false, pf[ac], false, vf, (short)0, acc, false, false);
            }
            o[db] = acc;
        }
    }

    for (int i = 0; i < 8; ++i) {
        float inv = 1.0f / lrow[i];
        int row = q0 + wave * 16 + mb + i;
        float* orow = out + ((size_t)b * N_ + row) * (size_t)D_;
        for (int db = 0; db < 8; ++db)
            orow[db * 16 + c] = o[db][i] * inv;
    }
}

extern "C" void kernel_launch(void* const* d_in, const int* in_sizes, int n_in,
                              void* d_out, int out_size, void* d_ws, size_t ws_size,
                              hipStream_t stream) {
    (void)in_sizes; (void)n_in; (void)out_size;
    const float* q = (const float*)d_in[0];
    const float* k = (const float*)d_in[1];
    const float* v = (const float*)d_in[2];
    float* o = (float*)d_out;

    if (d_ws != nullptr && ws_size >= WS_NEED) {
        char* ws = (char*)d_ws;
        hipLaunchKernelGGL(rbf_prep_kernel, dim3(B_ * (M_ / JT)), dim3(256), 0, stream,
                           k, v, ws);
        hipLaunchKernelGGL(rbf_attn_tdm_kernel, dim3(SPLITS * B_ * (N_ / QB)), dim3(256),
                           0, stream, q, ws);
        hipLaunchKernelGGL(rbf_combine_kernel, dim3((B_ * N_ * D_) / 256), dim3(256),
                           0, stream, ws, o);
    } else {
        hipLaunchKernelGGL(rbf_attn_fused_kernel, dim3(B_ * (N_ / QB)), dim3(256), 0, stream,
                           q, k, v, o);
    }
}
